// GraphTripleConv_76476187673051
// MI455X (gfx1250) — matrix operand order repaired and checked by reading, not compiled, not run on hardware
//
#include <hip/hip_runtime.h>
#include <hip/hip_bf16.h>
#include <stdint.h>

// ---------------------------------------------------------------------------
// GraphTripleConv on MI455X (gfx1250): bf16 WMMA GEMMs + fused BN/LeakyReLU
// ---------------------------------------------------------------------------

typedef __attribute__((ext_vector_type(16))) __bf16 v16bf;
typedef __attribute__((ext_vector_type(8)))  __bf16 bf16x8;
typedef __attribute__((ext_vector_type(2)))  __bf16 bf16x2;
typedef __attribute__((ext_vector_type(8)))  float  v8f;

#define NOBJ   50000
#define NTRIP  200000
#define DIN    128
#define HID    512
#define DOUT   128
#define ALPHA  0.2f
#define BN_EPS 1e-5f

__device__ __forceinline__ float lrelu(float v) { return v >= 0.f ? v : ALPHA * v; }

// ---------------------------------------------------------------------------
// Pack fp32 weight [K,N] into the per-lane WMMA B-fragment layout (bf16).
// Tile = (32 K x 16 N) = 1KB: dword index = tile*256 + lane*8 + j, holding
// bf16 pair W[k0][n], W[k0+1][n] with k0 = kt*32 + (lane>=16)*8 + (j>=4)*16 + 2(j&3), n = nt*16 + (lane&15).
// ---------------------------------------------------------------------------
__global__ void pack_w_kernel(const float* __restrict__ W, uint32_t* __restrict__ out,
                              int K, int N) {
    int idx = blockIdx.x * 256 + threadIdx.x;
    int total = (K / 2) * N;
    if (idx >= total) return;
    int j    = idx & 7;
    int lane = (idx >> 3) & 31;
    int tile = idx >> 8;
    int NT   = N >> 4;
    int nt   = tile % NT;
    int kt   = tile / NT;
    int k0   = kt * 32 + ((lane >> 4) << 3) + ((j >> 2) << 4) + 2 * (j & 3);
    int n    = nt * 16 + (lane & 15);
    union { bf16x2 v; uint32_t u; } cv;
    cv.v[0] = (__bf16)W[(size_t)k0 * N + n];
    cv.v[1] = (__bf16)W[(size_t)(k0 + 1) * N + n];
    out[idx] = cv.u;
}

// ---------------------------------------------------------------------------
// GEMM #1: gathered input.  cur_t[t] = [obj[s] | pred[t] | obj[o]]  (K=384)
// Out = bf16 pre-activation [T,512]; stats += column {sum, sum^2}.
// Block = 256 thr (8 waves); wave w -> ntile = blockIdx.y*8 + w; grid.y = 4.
// ---------------------------------------------------------------------------
__global__ __launch_bounds__(256) void gemm_gather_kernel(
    const float* __restrict__ obj, const float* __restrict__ pred,
    const int* __restrict__ edges, const uint32_t* __restrict__ Bp,
    const float* __restrict__ bias, __bf16* __restrict__ Out,
    float* __restrict__ stats) {
    constexpr int K = 3 * DIN;      // 384
    constexpr int KT = K / 32;      // 12
    constexpr int N = HID;          // 512
    constexpr int NT = N / 16;      // 32
    __shared__ __align__(16) __bf16 As[16][K + 8];
    __shared__ float shSum[128], shSq[128];
    __shared__ int sIdx[16], oIdx[16];

    int tid = threadIdx.x;
    int mtile = blockIdx.x, nblock = blockIdx.y;
    if (tid < 128) { shSum[tid] = 0.f; shSq[tid] = 0.f; }
    if (tid < 16) {
        int t = mtile * 16 + tid;
        sIdx[tid] = edges[2 * t];
        oIdx[tid] = edges[2 * t + 1];
    }
    __syncthreads();

    // stage gathered A tile: 16 x 384 bf16
    for (int c = 0; c < (16 * K) / 256; ++c) {
        int flat = c * 256 + tid;
        int r = flat / K;
        int k = flat - r * K;
        int t = mtile * 16 + r;
        float v;
        if (k < DIN)            v = obj[(size_t)sIdx[r] * DIN + k];
        else if (k < 2 * DIN)   v = pred[(size_t)t * DIN + (k - DIN)];
        else                    v = obj[(size_t)oIdx[r] * DIN + (k - 2 * DIN)];
        As[r][k] = (__bf16)v;
    }
    __syncthreads();

    int w = tid >> 5, lane = tid & 31;
    int ntile = nblock * 8 + w;
    int m = lane & 15;
    int kh = (lane >> 4) << 3;
    v8f acc = {};
#pragma unroll
    for (int kt = 0; kt < KT; ++kt) {
        union { v16bf v; bf16x8 h[2]; } a;
        a.h[0] = *reinterpret_cast<const bf16x8*>(&As[m][kt * 32 + kh]);
        a.h[1] = *reinterpret_cast<const bf16x8*>(&As[m][kt * 32 + 16 + kh]);
        v16bf b = reinterpret_cast<const v16bf*>(Bp)[(size_t)(kt * NT + ntile) * 32 + lane];
        acc = __builtin_amdgcn_wmma_f32_16x16x32_bf16(false, a.v, false, b,
                                                      (short)0, acc, false, false);
    }

    int nlocal = (w << 4) | (lane & 15);
    int col = nblock * 128 + nlocal;
    float bv = bias[col];
    float s = 0.f, q = 0.f;
    size_t row0 = (size_t)mtile * 16;
#pragma unroll
    for (int r = 0; r < 8; ++r) {
        int mr = r + ((lane >> 4) << 3);
        float pre = acc[r] + bv;
        s += pre; q += pre * pre;
        Out[(row0 + mr) * N + col] = (__bf16)pre;
    }
    atomicAdd(&shSum[nlocal], s);
    atomicAdd(&shSq[nlocal], q);
    __syncthreads();
    if (tid < 128) {
        atomicAdd(&stats[nblock * 128 + tid], shSum[tid]);
        atomicAdd(&stats[N + nblock * 128 + tid], shSq[tid]);
    }
}

// ---------------------------------------------------------------------------
// Generic K=512 GEMM: Out[M,N] = act(A) @ Bp + bias, bf16 in/out, f32 accum.
// BN_IN: apply y = lrelu(x*scale + shift) to A while staging (scale/shift in ssin).
// Also accumulates per-column {sum, sum^2} of the pre-activation into stats.
// ---------------------------------------------------------------------------
template <bool BN_IN>
__global__ __launch_bounds__(256) void gemm512_kernel(
    const __bf16* __restrict__ A, const uint32_t* __restrict__ Bp,
    const float* __restrict__ bias, const float* __restrict__ ssin,
    __bf16* __restrict__ Out, float* __restrict__ stats, int N) {
    constexpr int K = HID;          // 512
    constexpr int KT = K / 32;      // 16
    const int NT = N >> 4;
    __shared__ __align__(16) __bf16 As[16][K + 8];
    __shared__ float shSum[128], shSq[128];

    int tid = threadIdx.x;
    int mtile = blockIdx.x, nblock = blockIdx.y;
    if (tid < 128) { shSum[tid] = 0.f; shSq[tid] = 0.f; }

    {   // stage 16x512 A tile (8 bf16 per thread x 4 chunks), fused BN+lrelu
        const __bf16* Arow = A + (size_t)mtile * 16 * K;
#pragma unroll
        for (int c = 0; c < 4; ++c) {
            int flat = (tid + c * 256) << 3;
            int r = flat >> 9;
            int k = flat & (K - 1);
            bf16x8 v = *reinterpret_cast<const bf16x8*>(Arow + (size_t)r * K + k);
            if (BN_IN) {
#pragma unroll
                for (int i = 0; i < 8; ++i) {
                    float f = (float)v[i];
                    f = lrelu(f * ssin[k + i] + ssin[K + k + i]);
                    v[i] = (__bf16)f;
                }
            }
            *reinterpret_cast<bf16x8*>(&As[r][k]) = v;
        }
    }
    __syncthreads();

    int w = tid >> 5, lane = tid & 31;
    int ntile = nblock * 8 + w;
    int m = lane & 15;
    int kh = (lane >> 4) << 3;
    v8f acc = {};
#pragma unroll
    for (int kt = 0; kt < KT; ++kt) {
        union { v16bf v; bf16x8 h[2]; } a;
        a.h[0] = *reinterpret_cast<const bf16x8*>(&As[m][kt * 32 + kh]);
        a.h[1] = *reinterpret_cast<const bf16x8*>(&As[m][kt * 32 + 16 + kh]);
        v16bf b = reinterpret_cast<const v16bf*>(Bp)[(size_t)(kt * NT + ntile) * 32 + lane];
        acc = __builtin_amdgcn_wmma_f32_16x16x32_bf16(false, a.v, false, b,
                                                      (short)0, acc, false, false);
    }

    int nlocal = (w << 4) | (lane & 15);
    int col = nblock * 128 + nlocal;
    float bv = bias[col];
    float s = 0.f, q = 0.f;
    size_t row0 = (size_t)mtile * 16;
#pragma unroll
    for (int r = 0; r < 8; ++r) {
        int mr = r + ((lane >> 4) << 3);
        float pre = acc[r] + bv;
        s += pre; q += pre * pre;
        Out[(row0 + mr) * N + col] = (__bf16)pre;
    }
    atomicAdd(&shSum[nlocal], s);
    atomicAdd(&shSq[nlocal], q);
    __syncthreads();
    if (tid < 128) {
        atomicAdd(&stats[nblock * 128 + tid], shSum[tid]);
        atomicAdd(&stats[N + nblock * 128 + tid], shSq[tid]);
    }
}

// ---------------------------------------------------------------------------
// BN finalize: stats {sum, sumsq} -> {scale, shift}
// ---------------------------------------------------------------------------
__global__ void bn_finalize_kernel(const float* __restrict__ stats,
                                   const float* __restrict__ g,
                                   const float* __restrict__ be,
                                   float* __restrict__ ss, int N, float inv) {
    int i = blockIdx.x * 256 + threadIdx.x;
    if (i >= N) return;
    float mean = stats[i] * inv;
    float var  = stats[N + i] * inv - mean * mean;
    float sc   = g[i] * rsqrtf(var + BN_EPS);
    ss[i]      = sc;
    ss[N + i]  = be[i] - mean * sc;
}

// ---------------------------------------------------------------------------
// Scatter: new_t = bn_lrelu(new_t_pre); split into new_s / new_p / new_o;
// atomically pool new_s, new_o into pooled[O,512]; emit new_p; count edges.
// One block per triple, 256 threads sweep the 1152 columns (coalesced atomics).
// ---------------------------------------------------------------------------
__global__ __launch_bounds__(256) void scatter_kernel(
    const __bf16* __restrict__ newt, const float* __restrict__ ss,
    const int* __restrict__ edges, float* __restrict__ pooled,
    float* __restrict__ counts, float* __restrict__ outP) {
    constexpr int N = 2 * HID + DOUT;   // 1152
    int t = blockIdx.x;
    int s = edges[2 * t], o = edges[2 * t + 1];
    if (threadIdx.x == 0) {
        atomicAdd(&counts[s], 1.f);
        atomicAdd(&counts[o], 1.f);
    }
    const __bf16* row = newt + (size_t)t * N;
    for (int c = threadIdx.x; c < N; c += 256) {
        float v = lrelu((float)row[c] * ss[c] + ss[N + c]);
        if (c < HID)             atomicAdd(&pooled[(size_t)s * HID + c], v);
        else if (c < HID + DOUT) outP[(size_t)t * DOUT + (c - HID)] = v;
        else                     atomicAdd(&pooled[(size_t)o * HID + (c - HID - DOUT)], v);
    }
}

// pooled / clip(counts,1,O) -> bf16 GEMM input
__global__ void pooled_to_bf_kernel(const float* __restrict__ pooled,
                                    const float* __restrict__ counts,
                                    __bf16* __restrict__ out) {
    size_t i = (size_t)blockIdx.x * 256 + threadIdx.x;   // O*512 total
    int oi = (int)(i >> 9);
    float c = counts[oi];
    c = c < 1.f ? 1.f : (c > (float)NOBJ ? (float)NOBJ : c);
    out[i] = (__bf16)(pooled[i] / c);
}

// final: new_obj = bn_lrelu(out2_pre) -> fp32 output
__global__ void final_out_kernel(const __bf16* __restrict__ pre,
                                 const float* __restrict__ ss,
                                 float* __restrict__ out) {
    size_t i = (size_t)blockIdx.x * 256 + threadIdx.x;   // O*128 total
    int c = (int)(i & (DOUT - 1));
    float v = (float)pre[i] * ss[c] + ss[DOUT + c];
    out[i] = lrelu(v);
}

// ---------------------------------------------------------------------------
// Host driver (graph-capture safe: only stream ops)
// ---------------------------------------------------------------------------
static inline size_t alignup(size_t x) { return (x + 255) & ~(size_t)255; }

extern "C" void kernel_launch(void* const* d_in, const int* in_sizes, int n_in,
                              void* d_out, int out_size, void* d_ws, size_t ws_size,
                              hipStream_t stream) {
    (void)in_sizes; (void)n_in; (void)out_size;
    const float* obj   = (const float*)d_in[0];
    const float* pred  = (const float*)d_in[1];
    const int*   edges = (const int*)d_in[2];
    const float* W1a = (const float*)d_in[3],  *b1a = (const float*)d_in[4];
    const float* g1a = (const float*)d_in[5],  *be1a = (const float*)d_in[6];
    const float* W1b = (const float*)d_in[7],  *b1b = (const float*)d_in[8];
    const float* g1b = (const float*)d_in[9],  *be1b = (const float*)d_in[10];
    const float* W2a = (const float*)d_in[11], *b2a = (const float*)d_in[12];
    const float* g2a = (const float*)d_in[13], *be2a = (const float*)d_in[14];
    const float* W2b = (const float*)d_in[15], *b2b = (const float*)d_in[16];
    const float* g2b = (const float*)d_in[17], *be2b = (const float*)d_in[18];

    const int N1b = 2 * HID + DOUT;   // 1152
    char* ws = (char*)d_ws;
    size_t p = 0;
    auto take = [&](size_t bytes) { char* r = ws + p; p = alignup(p + bytes); return r; };

    uint32_t* W1a_p = (uint32_t*)take((size_t)384 * HID * 2);
    uint32_t* W1b_p = (uint32_t*)take((size_t)HID * N1b * 2);
    uint32_t* W2a_p = (uint32_t*)take((size_t)HID * HID * 2);
    uint32_t* W2b_p = (uint32_t*)take((size_t)HID * DOUT * 2);
    float* st1a = (float*)take((size_t)2 * HID * 4);
    float* st1b = (float*)take((size_t)2 * N1b * 4);
    float* st2a = (float*)take((size_t)2 * HID * 4);
    float* st2b = (float*)take((size_t)2 * DOUT * 4);
    float* ss1a = (float*)take((size_t)2 * HID * 4);
    float* ss1b = (float*)take((size_t)2 * N1b * 4);
    float* ss2a = (float*)take((size_t)2 * HID * 4);
    float* ss2b = (float*)take((size_t)2 * DOUT * 4);
    __bf16* h_pre   = (__bf16*)take((size_t)NTRIP * HID * 2);
    __bf16* nt_pre  = (__bf16*)take((size_t)NTRIP * N1b * 2);
    float*  pooled  = (float*)take((size_t)NOBJ * HID * 4);
    float*  counts  = (float*)take((size_t)NOBJ * 4);
    __bf16* pool_bf = (__bf16*)take((size_t)NOBJ * HID * 2);
    __bf16* h2_pre  = (__bf16*)take((size_t)NOBJ * HID * 2);
    __bf16* o2_pre  = (__bf16*)take((size_t)NOBJ * DOUT * 2);
    (void)ws_size;

    float* out_obj = (float*)d_out;                                 // [O,128]
    float* out_p   = (float*)d_out + (size_t)NOBJ * DOUT;           // [T,128]

    // zero accumulators (graph-capture legal)
    hipMemsetAsync(st1a, 0, 2 * HID * 4, stream);
    hipMemsetAsync(st1b, 0, 2 * N1b * 4, stream);
    hipMemsetAsync(st2a, 0, 2 * HID * 4, stream);
    hipMemsetAsync(st2b, 0, 2 * DOUT * 4, stream);
    hipMemsetAsync(pooled, 0, (size_t)NOBJ * HID * 4, stream);
    hipMemsetAsync(counts, 0, (size_t)NOBJ * 4, stream);

    // pack weights into WMMA-B fragment layout
    auto packs = [&](const float* W, uint32_t* Wp, int K, int N) {
        int dwords = (K / 2) * N;
        pack_w_kernel<<<(dwords + 255) / 256, 256, 0, stream>>>(W, Wp, K, N);
    };
    packs(W1a, W1a_p, 384, HID);
    packs(W1b, W1b_p, HID, N1b);
    packs(W2a, W2a_p, HID, HID);
    packs(W2b, W2b_p, HID, DOUT);

    const float invT = 1.f / (float)NTRIP;
    const float invO = 1.f / (float)NOBJ;

    // layer 1a: gather + GEMM [T,384]x[384,512]
    gemm_gather_kernel<<<dim3(NTRIP / 16, HID / 128), 256, 0, stream>>>(
        obj, pred, edges, W1a_p, b1a, h_pre, st1a);
    bn_finalize_kernel<<<(HID + 255) / 256, 256, 0, stream>>>(st1a, g1a, be1a, ss1a, HID, invT);

    // layer 1b: bn_lrelu(h) @ W1b  [T,512]x[512,1152]
    gemm512_kernel<true><<<dim3(NTRIP / 16, N1b / 128), 256, 0, stream>>>(
        h_pre, W1b_p, b1b, ss1a, nt_pre, st1b, N1b);
    bn_finalize_kernel<<<(N1b + 255) / 256, 256, 0, stream>>>(st1b, g1b, be1b, ss1b, N1b, invT);

    // scatter-mean pooling + emit new_p
    scatter_kernel<<<NTRIP, 256, 0, stream>>>(nt_pre, ss1b, edges, pooled, counts, out_p);
    pooled_to_bf_kernel<<<(NOBJ * HID) / 256, 256, 0, stream>>>(pooled, counts, pool_bf);

    // layer 2a: pooled @ W2a  [O,512]x[512,512]
    gemm512_kernel<false><<<dim3(NOBJ / 16, HID / 128), 256, 0, stream>>>(
        pool_bf, W2a_p, b2a, nullptr, h2_pre, st2a, HID);
    bn_finalize_kernel<<<(HID + 255) / 256, 256, 0, stream>>>(st2a, g2a, be2a, ss2a, HID, invO);

    // layer 2b: bn_lrelu(h2) @ W2b  [O,512]x[512,128]
    gemm512_kernel<true><<<dim3(NOBJ / 16, DOUT / 128), 256, 0, stream>>>(
        h2_pre, W2b_p, b2b, ss2a, o2_pre, st2b, DOUT);
    bn_finalize_kernel<<<(DOUT + 255) / 256, 256, 0, stream>>>(st2b, g2b, be2b, ss2b, DOUT, invO);

    // final bn_lrelu -> new_obj (fp32)
    final_out_kernel<<<(NOBJ * DOUT) / 256, 256, 0, stream>>>(o2_pre, ss2b, out_obj);
}
